// PointMAE_52003464020569
// MI455X (gfx1250) — matrix operand order repaired and checked by reading, not compiled, not run on hardware
//
#include <hip/hip_runtime.h>
#include <hip/hip_bf16.h>
#include <cstdint>
#include <cstddef>

// ---------------- constants ----------------
#define BATCH  32
#define NPTS   8192
#define NGRP   128
#define GK     32
#define LKEEP  32
#define DENC   768
#define DDEC   512
#define LE     12
#define LD     8

#define ST_SUM1 0
#define ST_SS1  64
#define ST_SUM2 128
#define ST_SS2  256
#define ST_SUM3 384
#define ST_SS3  1152
#define ST_LOSS 1920
#define ST_N    2048
#define BN_INV  (1.0f / 131072.0f)   // count = B*NG*K

typedef __attribute__((ext_vector_type(16))) _Float16 v16h;
typedef __attribute__((ext_vector_type(8)))  float    v8f;

// =====================================================================
// WMMA GEMM: C[M,N] = A[M,K] * W[N,K]^T + bias[N]   (optional ReLU)
// f32 inputs converted to f16 in-register; fp32 accumulate via
// v_wmma_f32_16x16x32_f16. 128 threads = 4 waves; block tile 64x64,
// wave tile 32x32 (2x2 WMMA frags).
// =====================================================================
__device__ inline v16h frag_a(const float* __restrict__ A, int row, int nrows,
                              int Kd, int k0, int half) {
  v16h f;
  if (row < nrows) {
    const float* p = A + (size_t)row * Kd + k0 + (half << 4);
#pragma unroll
    for (int i = 0; i < 8; ++i) f[i] = (_Float16)p[i];
#pragma unroll
    for (int i = 0; i < 8; ++i) f[i + 8] = (_Float16)p[i + 8];
  } else {
#pragma unroll
    for (int i = 0; i < 16; ++i) f[i] = (_Float16)0.0f;
  }
  return f;
}

__device__ inline v16h frag_b(const float* __restrict__ W, int row, int nrows,
                              int Kd, int k0, int half) {
  v16h f;
  if (row < nrows) {
    const float* p = W + (size_t)row * Kd + k0 + (half << 4);
#pragma unroll
    for (int i = 0; i < 16; ++i) f[i] = (_Float16)p[i];
  } else {
#pragma unroll
    for (int i = 0; i < 16; ++i) f[i] = (_Float16)0.0f;
  }
  return f;
}

__global__ __launch_bounds__(128)
void gemm_wmma_kernel(const float* __restrict__ A, const float* __restrict__ W,
                      const float* __restrict__ bias, float* __restrict__ C,
                      int M, int N, int Kd, int relu) {
  const int tid  = threadIdx.x;
  const int lane = tid & 31;
  const int wave = tid >> 5;
  const int half = (lane >> 4) & 1;
  const int r    = lane & 15;
  const int m0   = blockIdx.y * 64 + (wave >> 1) * 32;
  const int n0   = blockIdx.x * 64 + (wave & 1) * 32;

  v8f acc[2][2] = {};
  for (int k0 = 0; k0 < Kd; k0 += 32) {
    if (k0 + 32 < Kd) {  // speculative L2 prefetch of next K-slab
      __builtin_prefetch(A + (size_t)(m0 + r) * Kd + k0 + 32, 0, 1);
      __builtin_prefetch(W + (size_t)(n0 + r) * Kd + k0 + 32, 0, 1);
    }
    v16h a0 = frag_a(A, m0 + r,      M, Kd, k0, half);
    v16h a1 = frag_a(A, m0 + 16 + r, M, Kd, k0, half);
    v16h b0 = frag_b(W, n0 + r,      N, Kd, k0, half);
    v16h b1 = frag_b(W, n0 + 16 + r, N, Kd, k0, half);
    acc[0][0] = __builtin_amdgcn_wmma_f32_16x16x32_f16(false, a0, false, b0,
                                                       (short)0, acc[0][0], false, false);
    acc[0][1] = __builtin_amdgcn_wmma_f32_16x16x32_f16(false, a0, false, b1,
                                                       (short)0, acc[0][1], false, false);
    acc[1][0] = __builtin_amdgcn_wmma_f32_16x16x32_f16(false, a1, false, b0,
                                                       (short)0, acc[1][0], false, false);
    acc[1][1] = __builtin_amdgcn_wmma_f32_16x16x32_f16(false, a1, false, b1,
                                                       (short)0, acc[1][1], false, false);
  }
#pragma unroll
  for (int mi = 0; mi < 2; ++mi)
#pragma unroll
    for (int ni = 0; ni < 2; ++ni)
#pragma unroll
      for (int e = 0; e < 8; ++e) {
        int row = m0 + mi * 16 + (half << 3) + e;
        int col = n0 + ni * 16 + r;
        if (row < M && col < N) {
          float v = acc[mi][ni][e] + bias[col];
          if (relu) v = fmaxf(v, 0.0f);
          C[(size_t)row * N + col] = v;
        }
      }
}

// =====================================================================
// FPS: one block per batch; dmin resident in LDS; argmax w/ lowest-index
// tie-break (matches jnp.argmax).
// =====================================================================
__global__ __launch_bounds__(256)
void fps_kernel(const float* __restrict__ xyz, const int* __restrict__ init,
                int* __restrict__ cent_idx, float* __restrict__ cxyz) {
  const int b = blockIdx.x, tid = threadIdx.x;
  __shared__ float dmin[NPTS];
  __shared__ float rv[256];
  __shared__ int   ri[256];
  __shared__ int   s_far;
  const float* P = xyz + (size_t)b * NPTS * 3;
  for (int j = tid; j < NPTS; j += 256) dmin[j] = 1e10f;
  if (tid == 0) s_far = init[b];
  __syncthreads();
  for (int it = 0; it < NGRP; ++it) {
    int far = s_far;
    if (tid == 0) {
      cent_idx[b * NGRP + it] = far;
      cxyz[(size_t)(b * NGRP + it) * 3 + 0] = P[far * 3 + 0];
      cxyz[(size_t)(b * NGRP + it) * 3 + 1] = P[far * 3 + 1];
      cxyz[(size_t)(b * NGRP + it) * 3 + 2] = P[far * 3 + 2];
    }
    float cx = P[far * 3], cy = P[far * 3 + 1], cz = P[far * 3 + 2];
    float bv = -1.0f; int bi = 0;
    for (int j = tid; j < NPTS; j += 256) {
      float dx = P[j * 3] - cx, dy = P[j * 3 + 1] - cy, dz = P[j * 3 + 2] - cz;
      float d = fminf(dmin[j], dx * dx + dy * dy + dz * dz);
      dmin[j] = d;
      if (d > bv) { bv = d; bi = j; }
    }
    rv[tid] = bv; ri[tid] = bi;
    __syncthreads();
    for (int s = 128; s > 0; s >>= 1) {
      if (tid < s) {
        if (rv[tid + s] > rv[tid] ||
            (rv[tid + s] == rv[tid] && ri[tid + s] < ri[tid])) {
          rv[tid] = rv[tid + s]; ri[tid] = ri[tid + s];
        }
      }
      __syncthreads();
    }
    if (tid == 0) s_far = ri[0];
    __syncthreads();
  }
}

// =====================================================================
// KNN: one block per (b,g); 32 successive argmin rounds (== top_k order),
// emits centered neighborhoods gn[b,g,k,3].
// =====================================================================
__global__ __launch_bounds__(256)
void knn_kernel(const float* __restrict__ xyz, const float* __restrict__ cxyz,
                float* __restrict__ gn) {
  const int gi = blockIdx.x, b = gi >> 7, tid = threadIdx.x;
  __shared__ float dist[NPTS];
  __shared__ float rv[256];
  __shared__ int   ri[256];
  const float* P = xyz + (size_t)b * NPTS * 3;
  float cx = cxyz[(size_t)gi * 3], cy = cxyz[(size_t)gi * 3 + 1], cz = cxyz[(size_t)gi * 3 + 2];
  float c2 = cx * cx + cy * cy + cz * cz;
  for (int j = tid; j < NPTS; j += 256) {
    float x = P[j * 3], y = P[j * 3 + 1], z = P[j * 3 + 2];
    dist[j] = c2 + (x * x + y * y + z * z) - 2.0f * (cx * x + cy * y + cz * z);
  }
  __syncthreads();
  for (int kk = 0; kk < GK; ++kk) {
    float bv = 3.0e38f; int bi = NPTS;
    for (int j = tid; j < NPTS; j += 256) {
      float d = dist[j];
      if (d < bv) { bv = d; bi = j; }
    }
    rv[tid] = bv; ri[tid] = bi;
    __syncthreads();
    for (int s = 128; s > 0; s >>= 1) {
      if (tid < s) {
        if (rv[tid + s] < rv[tid] ||
            (rv[tid + s] == rv[tid] && ri[tid + s] < ri[tid])) {
          rv[tid] = rv[tid + s]; ri[tid] = ri[tid + s];
        }
      }
      __syncthreads();
    }
    if (tid == 0) {
      int j = ri[0];
      dist[j] = 3.0e38f;
      float* o = gn + ((size_t)gi * GK + kk) * 3;
      o[0] = P[j * 3 + 0] - cx; o[1] = P[j * 3 + 1] - cy; o[2] = P[j * 3 + 2] - cz;
    }
    __syncthreads();
  }
}

// =====================================================================
// Conv token pipeline (3->64->128->768, train-mode BN). Stats passes
// recompute activations per group in LDS to avoid a 400MB h3 buffer.
// =====================================================================
__global__ __launch_bounds__(256)
void zero_kernel(float* p, int n) {
  int i = blockIdx.x * blockDim.x + threadIdx.x;
  if (i < n) p[i] = 0.0f;
}

__global__ __launch_bounds__(256)
void conv_stats1_kernel(const float* __restrict__ gn, const float* __restrict__ cw1,
                        const float* __restrict__ cb1, float* __restrict__ stats) {
  const int gi = blockIdx.x, tid = threadIdx.x;
  __shared__ float s1[64], s2[64];
  if (tid < 64) { s1[tid] = 0.0f; s2[tid] = 0.0f; }
  __syncthreads();
  const float* G = gn + (size_t)gi * GK * 3;
  for (int idx = tid; idx < GK * 64; idx += 256) {
    int p = idx >> 6, ch = idx & 63;
    float v = cb1[ch] + G[p * 3 + 0] * cw1[ch * 3 + 0] +
              G[p * 3 + 1] * cw1[ch * 3 + 1] + G[p * 3 + 2] * cw1[ch * 3 + 2];
    atomicAdd(&s1[ch], v);
    atomicAdd(&s2[ch], v * v);
  }
  __syncthreads();
  if (tid < 64) {
    atomicAdd(&stats[ST_SUM1 + tid], s1[tid]);
    atomicAdd(&stats[ST_SS1 + tid], s2[tid]);
  }
}

__device__ inline void build_h1(const float* __restrict__ G, const float* __restrict__ cw1,
                                const float* __restrict__ cb1, const float* __restrict__ g1,
                                const float* __restrict__ be1, const float* __restrict__ stats,
                                float* __restrict__ h1, int tid) {
  for (int idx = tid; idx < GK * 64; idx += 256) {
    int p = idx >> 6, ch = idx & 63;
    float v = cb1[ch] + G[p * 3 + 0] * cw1[ch * 3 + 0] +
              G[p * 3 + 1] * cw1[ch * 3 + 1] + G[p * 3 + 2] * cw1[ch * 3 + 2];
    float m  = stats[ST_SUM1 + ch] * BN_INV;
    float va = stats[ST_SS1 + ch] * BN_INV - m * m;
    v = (v - m) * rsqrtf(va + 1e-5f) * g1[ch] + be1[ch];
    h1[idx] = fmaxf(v, 0.0f);
  }
}

__device__ inline void build_h2(const float* __restrict__ h1, const float* __restrict__ cw2,
                                const float* __restrict__ cb2, const float* __restrict__ g2,
                                const float* __restrict__ be2, const float* __restrict__ stats,
                                float* __restrict__ h2, int tid) {
  for (int idx = tid; idx < GK * 128; idx += 256) {
    int p = idx >> 7, ch = idx & 127;
    const float* hr = h1 + (p << 6);
    const float* wr = cw2 + ch * 64;
    float v = cb2[ch];
#pragma unroll 8
    for (int j = 0; j < 64; ++j) v += hr[j] * wr[j];
    float m  = stats[ST_SUM2 + ch] * BN_INV;
    float va = stats[ST_SS2 + ch] * BN_INV - m * m;
    v = (v - m) * rsqrtf(va + 1e-5f) * g2[ch] + be2[ch];
    h2[idx] = fmaxf(v, 0.0f);
  }
}

__global__ __launch_bounds__(256)
void conv_stats2_kernel(const float* __restrict__ gn, const float* __restrict__ cw1,
                        const float* __restrict__ cb1, const float* __restrict__ g1,
                        const float* __restrict__ be1, const float* __restrict__ cw2,
                        const float* __restrict__ cb2, float* __restrict__ stats) {
  const int gi = blockIdx.x, tid = threadIdx.x;
  __shared__ float h1[GK * 64];
  __shared__ float s1[128], s2[128];
  if (tid < 128) { s1[tid] = 0.0f; s2[tid] = 0.0f; }
  const float* G = gn + (size_t)gi * GK * 3;
  build_h1(G, cw1, cb1, g1, be1, stats, h1, tid);
  __syncthreads();
  for (int idx = tid; idx < GK * 128; idx += 256) {
    int p = idx >> 7, ch = idx & 127;
    const float* hr = h1 + (p << 6);
    const float* wr = cw2 + ch * 64;
    float v = cb2[ch];
#pragma unroll 8
    for (int j = 0; j < 64; ++j) v += hr[j] * wr[j];
    atomicAdd(&s1[ch], v);
    atomicAdd(&s2[ch], v * v);
  }
  __syncthreads();
  if (tid < 128) {
    atomicAdd(&stats[ST_SUM2 + tid], s1[tid]);
    atomicAdd(&stats[ST_SS2 + tid], s2[tid]);
  }
}

__global__ __launch_bounds__(256)
void conv_stats3_kernel(const float* __restrict__ gn, const float* __restrict__ cw1,
                        const float* __restrict__ cb1, const float* __restrict__ g1,
                        const float* __restrict__ be1, const float* __restrict__ cw2,
                        const float* __restrict__ cb2, const float* __restrict__ g2,
                        const float* __restrict__ be2, const float* __restrict__ cw3,
                        const float* __restrict__ cb3, float* __restrict__ stats) {
  const int gi = blockIdx.x, tid = threadIdx.x;
  __shared__ float h1[GK * 64];
  __shared__ float h2[GK * 128];
  const float* G = gn + (size_t)gi * GK * 3;
  build_h1(G, cw1, cb1, g1, be1, stats, h1, tid);
  __syncthreads();
  build_h2(h1, cw2, cb2, g2, be2, stats, h2, tid);
  __syncthreads();
  for (int ch = tid; ch < DENC; ch += 256) {
    const float* wr = cw3 + ch * 128;
    float s = 0.0f, ss = 0.0f;
    for (int p = 0; p < GK; ++p) {
      const float* hr = h2 + (p << 7);
      float v = cb3[ch];
#pragma unroll 8
      for (int j = 0; j < 128; ++j) v += hr[j] * wr[j];
      s += v; ss += v * v;
    }
    atomicAdd(&stats[ST_SUM3 + ch], s);
    atomicAdd(&stats[ST_SS3 + ch], ss);
  }
}

// conv3 + BN3 + maxpool(K) + positional embed -> x tokens (B,NG,768)
__global__ __launch_bounds__(256)
void token_kernel(const float* __restrict__ gn, const float* __restrict__ cw1,
                  const float* __restrict__ cb1, const float* __restrict__ g1,
                  const float* __restrict__ be1, const float* __restrict__ cw2,
                  const float* __restrict__ cb2, const float* __restrict__ g2,
                  const float* __restrict__ be2, const float* __restrict__ cw3,
                  const float* __restrict__ cb3, const float* __restrict__ g3,
                  const float* __restrict__ be3, const float* __restrict__ stats,
                  const float* __restrict__ cxyz, const float* __restrict__ pos_w,
                  const float* __restrict__ pos_b, float* __restrict__ x) {
  const int gi = blockIdx.x, tid = threadIdx.x;
  __shared__ float h1[GK * 64];
  __shared__ float h2[GK * 128];
  const float* G = gn + (size_t)gi * GK * 3;
  build_h1(G, cw1, cb1, g1, be1, stats, h1, tid);
  __syncthreads();
  build_h2(h1, cw2, cb2, g2, be2, stats, h2, tid);
  __syncthreads();
  float cx = cxyz[(size_t)gi * 3], cy = cxyz[(size_t)gi * 3 + 1], cz = cxyz[(size_t)gi * 3 + 2];
  for (int ch = tid; ch < DENC; ch += 256) {
    const float* wr = cw3 + ch * 128;
    float m  = stats[ST_SUM3 + ch] * BN_INV;
    float va = stats[ST_SS3 + ch] * BN_INV - m * m;
    float sc = rsqrtf(va + 1e-5f) * g3[ch];
    float mx = -3.0e38f;
    for (int p = 0; p < GK; ++p) {
      const float* hr = h2 + (p << 7);
      float v = cb3[ch];
#pragma unroll 8
      for (int j = 0; j < 128; ++j) v += hr[j] * wr[j];
      v = (v - m) * sc + be3[ch];
      mx = fmaxf(mx, v);
    }
    float pos = cx * pos_w[ch * 3] + cy * pos_w[ch * 3 + 1] + cz * pos_w[ch * 3 + 2] + pos_b[ch];
    x[(size_t)gi * DENC + ch] = mx + pos;
  }
}

// =====================================================================
// shuffle bookkeeping
// =====================================================================
__global__ __launch_bounds__(256)
void restore_kernel(const int* __restrict__ shuf, int* __restrict__ restore) {
  int i = blockIdx.x * blockDim.x + threadIdx.x;
  if (i < BATCH * NGRP) {
    int b = i >> 7, p = i & 127;
    restore[b * NGRP + shuf[i]] = p;
  }
}

__global__ __launch_bounds__(256)
void gather_keep_kernel(const float* __restrict__ x, const int* __restrict__ shuf,
                        float* __restrict__ xk) {
  int row = blockIdx.x;               // b*32+s
  int b = row >> 5, s = row & 31;
  int src = shuf[b * NGRP + s];
  const float* in = x + ((size_t)b * NGRP + src) * DENC;
  float* out = xk + (size_t)row * DENC;
  for (int c = threadIdx.x; c < DENC; c += blockDim.x) out[c] = in[c];
}

__global__ __launch_bounds__(256)
void mask_kernel(const int* __restrict__ restore, float* __restrict__ mask) {
  int i = blockIdx.x * blockDim.x + threadIdx.x;
  if (i < BATCH * NGRP) mask[i] = (restore[i] >= LKEEP) ? 1.0f : 0.0f;
}

// =====================================================================
// attention: one block per (b,h); K/V staged in LDS; streaming softmax
// =====================================================================
template <int DH>
__global__ __launch_bounds__(128)
void attn_kernel(const float* __restrict__ qkv, float* __restrict__ out,
                 int S, int D, int nh) {
  extern __shared__ float lds[];
  float* kl = lds;
  float* vl = lds + S * DH;
  const int bh = blockIdx.x, b = bh / nh, h = bh % nh, tid = threadIdx.x;
  const float* base = qkv + (size_t)b * S * 3 * D + (size_t)h * DH;
  for (int idx = tid; idx < S * DH; idx += blockDim.x) {
    int s = idx / DH, c = idx % DH;
    kl[idx] = base[(size_t)s * 3 * D + D + c];
    vl[idx] = base[(size_t)s * 3 * D + 2 * D + c];
  }
  __syncthreads();
  if (tid < S) {
    float q[DH];
#pragma unroll
    for (int c = 0; c < DH; ++c) q[c] = base[(size_t)tid * 3 * D + c];
    const float scale = rsqrtf((float)DH);
    float m = -3.0e38f;
    for (int j = 0; j < S; ++j) {
      float s = 0.0f;
#pragma unroll
      for (int c = 0; c < DH; ++c) s += q[c] * kl[j * DH + c];
      m = fmaxf(m, s * scale);
    }
    float acc[DH];
#pragma unroll
    for (int c = 0; c < DH; ++c) acc[c] = 0.0f;
    float den = 0.0f;
    for (int j = 0; j < S; ++j) {
      float s = 0.0f;
#pragma unroll
      for (int c = 0; c < DH; ++c) s += q[c] * kl[j * DH + c];
      float e = __expf(s * scale - m);
      den += e;
#pragma unroll
      for (int c = 0; c < DH; ++c) acc[c] += e * vl[j * DH + c];
    }
    float inv = 1.0f / den;
    float* o = out + ((size_t)b * S + tid) * D + (size_t)h * DH;
#pragma unroll
    for (int c = 0; c < DH; ++c) o[c] = acc[c] * inv;
  }
}

// =====================================================================
// residual add + LayerNorm (one block per row)
// =====================================================================
__global__ __launch_bounds__(256)
void add_ln_kernel(const float* __restrict__ x, const float* __restrict__ r,
                   const float* __restrict__ g, const float* __restrict__ be,
                   float* __restrict__ out, int D) {
  const int row = blockIdx.x, tid = threadIdx.x;
  __shared__ float buf[DENC];
  __shared__ float red[256];
  const float* X = x + (size_t)row * D;
  const float* R = r + (size_t)row * D;
  float s = 0.0f;
  for (int c = tid; c < D; c += 256) {
    float t = X[c] + R[c];
    buf[c] = t;
    s += t;
  }
  red[tid] = s; __syncthreads();
  for (int st = 128; st > 0; st >>= 1) {
    if (tid < st) red[tid] += red[tid + st];
    __syncthreads();
  }
  float mean = red[0] / (float)D;
  __syncthreads();
  float s2 = 0.0f;
  for (int c = tid; c < D; c += 256) {
    float t = buf[c] - mean;
    s2 += t * t;
  }
  red[tid] = s2; __syncthreads();
  for (int st = 128; st > 0; st >>= 1) {
    if (tid < st) red[tid] += red[tid + st];
    __syncthreads();
  }
  float k = rsqrtf(red[0] / (float)D + 1e-5f);
  for (int c = tid; c < D; c += 256)
    out[(size_t)row * D + c] = (buf[c] - mean) * k * g[c] + be[c];
}

// =====================================================================
// decoder sequence assembly: unshuffle + mask token + decoder pos embed
// =====================================================================
__global__ __launch_bounds__(256)
void dec_embed_kernel(const float* __restrict__ y, const float* __restrict__ mtok,
                      const int* __restrict__ restore, const float* __restrict__ cxyz,
                      const float* __restrict__ dpw, const float* __restrict__ dpb,
                      float* __restrict__ yd) {
  const int gi = blockIdx.x, b = gi >> 7, tid = threadIdx.x;
  int p = restore[gi];
  const float* src = (p < LKEEP) ? (y + ((size_t)b * LKEEP + p) * DDEC) : mtok;
  float cx = cxyz[(size_t)gi * 3], cy = cxyz[(size_t)gi * 3 + 1], cz = cxyz[(size_t)gi * 3 + 2];
  float* o = yd + (size_t)gi * DDEC;
  for (int c = tid; c < DDEC; c += 256)
    o[c] = src[c] + cx * dpw[c * 3] + cy * dpw[c * 3 + 1] + cz * dpw[c * 3 + 2] + dpb[c];
}

// =====================================================================
// masked reconstruction loss
// =====================================================================
__global__ __launch_bounds__(128)
void loss_accum_kernel(const float* __restrict__ pred, const float* __restrict__ gn,
                       const float* __restrict__ mask, float* __restrict__ acc) {
  const int gi = blockIdx.x, tid = threadIdx.x;
  __shared__ float red[128];
  const float* P = pred + (size_t)gi * 96;
  const float* T = gn + (size_t)gi * 96;
  float s = 0.0f;
  for (int e = tid; e < 96; e += 128) {
    float d = P[e] - T[e];
    s += d * d;
  }
  red[tid] = s; __syncthreads();
  for (int st = 64; st > 0; st >>= 1) {
    if (tid < st) red[tid] += red[tid + st];
    __syncthreads();
  }
  if (tid == 0) {
    float m = mask[gi];
    atomicAdd(&acc[0], (red[0] / 96.0f) * m);
    atomicAdd(&acc[1], m);
  }
}

__global__ void loss_final_kernel(const float* __restrict__ acc, float* __restrict__ out) {
  if (threadIdx.x == 0 && blockIdx.x == 0) out[0] = acc[0] / acc[1];
}

// =====================================================================
// orchestration
// =====================================================================
extern "C" void kernel_launch(void* const* d_in, const int* in_sizes, int n_in,
                              void* d_out, int out_size, void* d_ws, size_t ws_size,
                              hipStream_t stream) {
  const float* xyz      = (const float*)d_in[0];
  const int*   fps_init = (const int*)d_in[1];
  const int*   shuf     = (const int*)d_in[2];
  const float *cw1 = (const float*)d_in[3],  *cb1 = (const float*)d_in[4];
  const float *g1  = (const float*)d_in[5],  *be1 = (const float*)d_in[6];
  const float *cw2 = (const float*)d_in[7],  *cb2 = (const float*)d_in[8];
  const float *g2  = (const float*)d_in[9],  *be2 = (const float*)d_in[10];
  const float *cw3 = (const float*)d_in[11], *cb3 = (const float*)d_in[12];
  const float *g3  = (const float*)d_in[13], *be3 = (const float*)d_in[14];
  const float *pos_w = (const float*)d_in[15], *pos_b = (const float*)d_in[16];
  const float *eqw = (const float*)d_in[17], *eqb = (const float*)d_in[18];
  const float *eow = (const float*)d_in[19], *eob = (const float*)d_in[20];
  const float *el1g = (const float*)d_in[21], *el1b = (const float*)d_in[22];
  const float *ef1w = (const float*)d_in[23], *ef1b = (const float*)d_in[24];
  const float *ef2w = (const float*)d_in[25], *ef2b = (const float*)d_in[26];
  const float *el2g = (const float*)d_in[27], *el2b = (const float*)d_in[28];
  const float *dew = (const float*)d_in[29], *deb = (const float*)d_in[30];
  const float *mtok = (const float*)d_in[31];
  const float *dpw = (const float*)d_in[32], *dpb = (const float*)d_in[33];
  const float *dqw = (const float*)d_in[34], *dqb = (const float*)d_in[35];
  const float *dow = (const float*)d_in[36], *dob = (const float*)d_in[37];
  const float *dl1g = (const float*)d_in[38], *dl1b = (const float*)d_in[39];
  const float *df1w = (const float*)d_in[40], *df1b = (const float*)d_in[41];
  const float *df2w = (const float*)d_in[42], *df2b = (const float*)d_in[43];
  const float *dl2g = (const float*)d_in[44], *dl2b = (const float*)d_in[45];
  const float *prw = (const float*)d_in[46], *prb = (const float*)d_in[47];

  // workspace arena (float offsets)
  float* w = (float*)d_ws;
  constexpr size_t OFF_CENT  = 0;                        // int[4096]
  constexpr size_t OFF_CXYZ  = OFF_CENT  + 4096;         // 12288
  constexpr size_t OFF_REST  = OFF_CXYZ  + 12288;        // int[4096]
  constexpr size_t OFF_GN    = OFF_REST  + 4096;         // 393216
  constexpr size_t OFF_STATS = OFF_GN    + 393216;       // 2048
  constexpr size_t OFF_TOK   = OFF_STATS + ST_N;         // 4096*768
  constexpr size_t OFF_SA    = OFF_TOK   + 3145728;      // 1024*768
  constexpr size_t OFF_SB    = OFF_SA    + 786432;       // 1024*768
  constexpr size_t OFF_QKV   = OFF_SB    + 786432;       // max(1024*2304, 4096*1536)
  constexpr size_t OFF_FFN   = OFF_QKV   + 6291456;      // max(1024*3072, 4096*2048)
  constexpr size_t OFF_DA    = OFF_FFN   + 8388608;      // 4096*512
  constexpr size_t OFF_DB    = OFF_DA    + 2097152;      // 4096*512

  int*   CENT  = (int*)(w + OFF_CENT);
  float* CXYZ  = w + OFF_CXYZ;
  int*   REST  = (int*)(w + OFF_REST);
  float* GN    = w + OFF_GN;
  float* STATS = w + OFF_STATS;
  float* TOK   = w + OFF_TOK;
  float* SA    = w + OFF_SA;
  float* SB    = w + OFF_SB;
  float* QKV   = w + OFF_QKV;
  float* FFN   = w + OFF_FFN;
  float* DA    = w + OFF_DA;
  float* DB    = w + OFF_DB;

  float* out     = (float*)d_out;
  float* predOut = out + 1;
  float* maskOut = out + 1 + BATCH * NGRP * 96;

  auto gemm = [&](const float* A, const float* W, const float* bias, float* C,
                  int M, int N, int Kd, int relu) {
    dim3 grid((N + 63) / 64, (M + 63) / 64);
    gemm_wmma_kernel<<<grid, 128, 0, stream>>>(A, W, bias, C, M, N, Kd, relu);
  };

  // ---- grouping ----
  zero_kernel<<<(ST_N + 255) / 256, 256, 0, stream>>>(STATS, ST_N);
  fps_kernel<<<BATCH, 256, 0, stream>>>(xyz, fps_init, CENT, CXYZ);
  knn_kernel<<<BATCH * NGRP, 256, 0, stream>>>(xyz, CXYZ, GN);

  // ---- token embed (conv + train-BN + maxpool + pos) ----
  conv_stats1_kernel<<<BATCH * NGRP, 256, 0, stream>>>(GN, cw1, cb1, STATS);
  conv_stats2_kernel<<<BATCH * NGRP, 256, 0, stream>>>(GN, cw1, cb1, g1, be1, cw2, cb2, STATS);
  conv_stats3_kernel<<<BATCH * NGRP, 256, 0, stream>>>(GN, cw1, cb1, g1, be1, cw2, cb2,
                                                       g2, be2, cw3, cb3, STATS);
  token_kernel<<<BATCH * NGRP, 256, 0, stream>>>(GN, cw1, cb1, g1, be1, cw2, cb2, g2, be2,
                                                 cw3, cb3, g3, be3, STATS, CXYZ,
                                                 pos_w, pos_b, TOK);

  // ---- masking bookkeeping ----
  restore_kernel<<<(BATCH * NGRP + 255) / 256, 256, 0, stream>>>(shuf, REST);
  gather_keep_kernel<<<BATCH * LKEEP, 256, 0, stream>>>(TOK, shuf, SA);

  // ---- encoder: 12 layers, D=768, 12 heads, S=32, M=1024 ----
  const int Me = BATCH * LKEEP;
  for (int i = 0; i < LE; ++i) {
    const float* qw = eqw + (size_t)i * 3 * DENC * DENC;
    const float* qb = eqb + (size_t)i * 3 * DENC;
    const float* ow = eow + (size_t)i * DENC * DENC;
    const float* ob = eob + (size_t)i * DENC;
    const float* f1w = ef1w + (size_t)i * 4 * DENC * DENC;
    const float* f1b = ef1b + (size_t)i * 4 * DENC;
    const float* f2w = ef2w + (size_t)i * DENC * 4 * DENC;
    const float* f2b = ef2b + (size_t)i * DENC;
    gemm(SA, qw, qb, QKV, Me, 3 * DENC, DENC, 0);
    attn_kernel<64><<<BATCH * 12, 32, 32 * 64 * 2 * sizeof(float), stream>>>(
        QKV, SB, LKEEP, DENC, 12);
    gemm(SB, ow, ob, FFN, Me, DENC, DENC, 0);
    add_ln_kernel<<<Me, 256, 0, stream>>>(FFN, SA, el1g + (size_t)i * DENC,
                                          el1b + (size_t)i * DENC, SB, DENC);
    gemm(SB, f1w, f1b, FFN, Me, 4 * DENC, DENC, 1);
    gemm(FFN, f2w, f2b, QKV, Me, DENC, 4 * DENC, 0);
    add_ln_kernel<<<Me, 256, 0, stream>>>(QKV, SB, el2g + (size_t)i * DENC,
                                          el2b + (size_t)i * DENC, SA, DENC);
  }

  // ---- decoder embed + unshuffle ----
  gemm(SA, dew, deb, SB, Me, DDEC, DENC, 0);
  dec_embed_kernel<<<BATCH * NGRP, 256, 0, stream>>>(SB, mtok, REST, CXYZ, dpw, dpb, DA);

  // ---- decoder: 8 layers, D=512, 16 heads, S=128, M=4096 ----
  const int Md = BATCH * NGRP;
  for (int i = 0; i < LD; ++i) {
    const float* qw = dqw + (size_t)i * 3 * DDEC * DDEC;
    const float* qb = dqb + (size_t)i * 3 * DDEC;
    const float* ow = dow + (size_t)i * DDEC * DDEC;
    const float* ob = dob + (size_t)i * DDEC;
    const float* f1w = df1w + (size_t)i * 4 * DDEC * DDEC;
    const float* f1b = df1b + (size_t)i * 4 * DDEC;
    const float* f2w = df2w + (size_t)i * DDEC * 4 * DDEC;
    const float* f2b = df2b + (size_t)i * DDEC;
    gemm(DA, qw, qb, QKV, Md, 3 * DDEC, DDEC, 0);
    attn_kernel<32><<<BATCH * 16, 128, 128 * 32 * 2 * sizeof(float), stream>>>(
        QKV, DB, NGRP, DDEC, 16);
    gemm(DB, ow, ob, FFN, Md, DDEC, DDEC, 0);
    add_ln_kernel<<<Md, 256, 0, stream>>>(FFN, DA, dl1g + (size_t)i * DDEC,
                                          dl1b + (size_t)i * DDEC, DB, DDEC);
    gemm(DB, f1w, f1b, FFN, Md, 4 * DDEC, DDEC, 1);
    gemm(FFN, f2w, f2b, QKV, Md, DDEC, 4 * DDEC, 0);
    add_ln_kernel<<<Md, 256, 0, stream>>>(QKV, DB, dl2g + (size_t)i * DDEC,
                                          dl2b + (size_t)i * DDEC, DA, DDEC);
  }

  // ---- prediction head + loss ----
  gemm(DA, prw, prb, predOut, Md, GK * 3, DDEC, 0);
  mask_kernel<<<(BATCH * NGRP + 255) / 256, 256, 0, stream>>>(REST, maskOut);
  loss_accum_kernel<<<BATCH * NGRP, 128, 0, stream>>>(predOut, GN, maskOut, STATS + ST_LOSS);
  loss_final_kernel<<<1, 32, 0, stream>>>(STATS + ST_LOSS, out);
}
